// DocLSTM_34832184771187
// MI455X (gfx1250) — matrix operand order repaired and checked by reading, not compiled, not run on hardware
//
#include <hip/hip_runtime.h>
#include <hip/hip_bf16.h>
#include <cstdint>
#include <cstddef>

// ---------------------------------------------------------------------------
// Model dims (from reference)
// ---------------------------------------------------------------------------
#define MEM      100
#define IN_DIM   300
#define N_SENT   512
#define N_WORD   64
#define N_HEAD   8
#define DMODEL   200            // 2*MEM
#define NGATE    400            // 4*MEM
#define NTILES_G 25             // 400/16
#define KC_IH    10             // ceil(300/32) -> K padded to 320
#define KC_HH    4              // ceil(100/32) -> K padded to 128
#define NSEQ     513            // 512 body sentences + 1 headline
#define SEQ_TILES 33            // ceil(513/16)

typedef __attribute__((ext_vector_type(16))) _Float16 v16h;
typedef __attribute__((ext_vector_type(8)))  float    v8f;

__device__ __forceinline__ v8f wmma_f16(v16h a, v16h b, v8f c) {
  // D = A(16x32 f16) * B(32x16 f16) + C(16x16 f32)
  return __builtin_amdgcn_wmma_f32_16x16x32_f16(false, a, false, b, (short)0, c,
                                                false, false);
}

// 16-bit A-matrix 16x32 K mapping: lane half i, lane-high bit hi -> K index.
__device__ __forceinline__ int kmapA(int i, int hi) {
  return (i < 8) ? (i + hi * 8) : (i - 8 + 16 + hi * 8);
}
__device__ __forceinline__ float sigf(float x) { return 1.f / (1.f + __expf(-x)); }

// ---------------------------------------------------------------------------
// Pack LSTM weights into WMMA B-fragment layout:
//   pk[((dir*25 + nt)*KC + kc)*512 + lane*16 + i] = W[nt*16 + (lane&15)][kc*32 + kmapA(i, lane>>4)]
// (B 32x16 f16: lanes 0-15 hold K=0..7,16..23, lanes 16-31 hold K=8..15,24..31; N = lane&15)
// ---------------------------------------------------------------------------
__global__ void k_pack_w(const float* __restrict__ wfwd, const float* __restrict__ wbwd,
                         _Float16* __restrict__ pk, int K, int KC) {
  size_t total = (size_t)2 * NTILES_G * KC * 512;
  size_t idx = (size_t)blockIdx.x * blockDim.x + threadIdx.x;
  if (idx >= total) return;
  int i  = (int)(idx & 15);
  int l  = (int)((idx >> 4) & 31);
  size_t rem = idx >> 9;
  int kc = (int)(rem % (size_t)KC); rem /= (size_t)KC;
  int nt = (int)(rem % NTILES_G);
  int dir = (int)(rem / NTILES_G);
  int n = nt * 16 + (l & 15);
  int k = kc * 32 + kmapA(i, l >> 4);
  const float* w = dir ? wbwd : wfwd;
  float v = (k < K) ? w[(size_t)n * K + k] : 0.f;
  pk[idx] = (_Float16)v;
}

__global__ void k_bias(const float* __restrict__ bif, const float* __restrict__ bhf,
                       const float* __restrict__ bib, const float* __restrict__ bhb,
                       float* __restrict__ bias_c) {
  int idx = blockIdx.x * 256 + threadIdx.x;
  if (idx >= 2 * NGATE) return;
  int dir = idx / NGATE, n = idx % NGATE;
  bias_c[idx] = dir ? (bib[n] + bhb[n]) : (bif[n] + bhf[n]);
}

// ---------------------------------------------------------------------------
// Embedding gather straight into WMMA A-fragment layout:
//   xfrag[((tile*64 + t)*10 + kc)*512 + lane*16 + i] = emb[tok(seq,t)][kc*32 + kmapA(i,hi)]
// seq = tile*16 + (lane&15); seq==512 is the headline; >512 zero-pad.
// ---------------------------------------------------------------------------
__global__ void k_gather(const int* __restrict__ head_tok, const int* __restrict__ body_tok,
                         const float* __restrict__ emb, _Float16* __restrict__ xfrag) {
  size_t total = (size_t)SEQ_TILES * N_WORD * KC_IH * 512;
  size_t idx = (size_t)blockIdx.x * blockDim.x + threadIdx.x;
  if (idx >= total) return;
  int i  = (int)(idx & 15);
  int l  = (int)((idx >> 4) & 31);
  size_t rem = idx >> 9;
  int kc = (int)(rem % KC_IH);
  size_t tt = rem / KC_IH;
  int t = (int)(tt & 63);
  int tile = (int)(tt >> 6);
  int m = l & 15, hi = l >> 4;
  int seq = tile * 16 + m;
  int k = kc * 32 + kmapA(i, hi);
  float v = 0.f;
  if (seq < NSEQ && k < IN_DIM) {
    int tok = (seq == N_SENT) ? head_tok[t] : body_tok[seq * N_WORD + t];
    v = emb[(size_t)tok * IN_DIM + k];
  }
  xfrag[idx] = (_Float16)v;
}

// ---------------------------------------------------------------------------
// Batched BiLSTM: one wave handles a 16-sequence tile for one direction,
// looping 64 timesteps.  Per step: gates[16,400] = Xt@Wih^T + H@Whh^T + b
// via 25 x (10+4) v_wmma_f32_16x16x32_f16, then VALU gate nonlinearity with
// C-state in LDS and H written back in A-fragment layout for the next step.
// ---------------------------------------------------------------------------
__global__ void __launch_bounds__(32)
k_lstm(const _Float16* __restrict__ xfrag,
       const _Float16* __restrict__ pk_ih,
       const _Float16* __restrict__ pk_hh,
       const float* __restrict__ bias_c,
       float* __restrict__ sent_enc /* [NSEQ][200] */) {
  __shared__ v16h  hfragv[KC_HH * 32];     // recurrent H in A-fragment layout
  __shared__ float cst[16 * MEM];          // cell state
  __shared__ float gate[16 * NGATE];       // gate pre-activations
  _Float16* hfrag = (_Float16*)hfragv;

  int tile = blockIdx.x;       // 0..32
  int dir  = blockIdx.y;       // 0 = forward, 1 = backward (x reversed)
  int lane = threadIdx.x;
  int n_lo = lane & 15, hi = lane >> 4;

  for (int j = lane; j < KC_HH * 32 * 16; j += 32) hfrag[j] = (_Float16)0.f;
  for (int j = lane; j < 16 * MEM; j += 32) cst[j] = 0.f;

  const _Float16* wih = pk_ih + (size_t)dir * NTILES_G * KC_IH * 512;
  const _Float16* whh = pk_hh + (size_t)dir * NTILES_G * KC_HH * 512;
  const float* bc = bias_c + dir * NGATE;

  for (int step = 0; step < N_WORD; ++step) {
    int t = dir ? (N_WORD - 1 - step) : step;
    const _Float16* xb = xfrag + ((size_t)(tile * N_WORD + t)) * KC_IH * 512;

    for (int nt = 0; nt < NTILES_G; ++nt) {
      __builtin_prefetch((const void*)(wih + ((size_t)((nt + 1) % NTILES_G) * KC_IH) * 512), 0, 1);
      float bv0 = bc[nt * 16 + n_lo];
      v8f acc = {bv0, bv0, bv0, bv0, bv0, bv0, bv0, bv0};
#pragma unroll
      for (int kc = 0; kc < KC_IH; ++kc) {
        v16h a = *(const v16h*)(xb + (size_t)kc * 512 + lane * 16);
        v16h b = *(const v16h*)(wih + ((size_t)nt * KC_IH + kc) * 512 + lane * 16);
        acc = wmma_f16(a, b, acc);
      }
#pragma unroll
      for (int kc = 0; kc < KC_HH; ++kc) {
        v16h a = hfragv[kc * 32 + lane];
        v16h b = *(const v16h*)(whh + ((size_t)nt * KC_HH + kc) * 512 + lane * 16);
        acc = wmma_f16(a, b, acc);
      }
      // C/D layout: vgpr r, lane l -> m = r + 8*(l>>4), n = l&15
#pragma unroll
      for (int r = 0; r < 8; ++r) {
        int m = r + 8 * hi;
        gate[m * NGATE + nt * 16 + n_lo] = acc[r];
      }
    }
    // elementwise LSTM update (torch gate order i, f, g, o)
    for (int u = lane; u < MEM; u += 32) {
      int kc = u >> 5, kk = u & 31;
      int fhi = (kk >> 3) & 1;
      int fi = (kk & 7) + ((kk >= 16) ? 8 : 0);
      for (int m = 0; m < 16; ++m) {
        float ig = gate[m * NGATE + u];
        float fg = gate[m * NGATE + MEM + u];
        float gg = gate[m * NGATE + 2 * MEM + u];
        float og = gate[m * NGATE + 3 * MEM + u];
        float c = cst[m * MEM + u];
        c = sigf(fg) * c + sigf(ig) * tanhf(gg);
        float h = sigf(og) * tanhf(c);
        cst[m * MEM + u] = c;
        hfrag[((size_t)kc * 32 + (m + 16 * fhi)) * 16 + fi] = (_Float16)h;
      }
    }
  }
  // final hidden -> sent_enc[seq][dir*100 + u]
  for (int u = lane; u < MEM; u += 32) {
    int kc = u >> 5, kk = u & 31;
    int fhi = (kk >> 3) & 1;
    int fi = (kk & 7) + ((kk >= 16) ? 8 : 0);
    for (int m = 0; m < 16; ++m) {
      int seq = tile * 16 + m;
      if (seq < NSEQ) {
        sent_enc[(size_t)seq * DMODEL + dir * MEM + u] =
            (float)hfrag[((size_t)kc * 32 + (m + 16 * fhi)) * 16 + fi];
      }
    }
  }
}

// ---------------------------------------------------------------------------
// Generic one-wave-per-16x16-tile f16 WMMA GEMM with LDS fragment staging.
//   C[m,n] = scale * (sum_k A[m,k]*B[k,n] + bias[n])
//   transB: B[k,n] = Braw[n*ldb + k]   (else Braw[k*ldb + n])
// Batched over blockIdx.z with 64-bit strides (stride 0 = shared operand).
// ---------------------------------------------------------------------------
__global__ void __launch_bounds__(32)
k_gemm16(const _Float16* __restrict__ A, int lda, long long sA,
         const _Float16* __restrict__ B, int ldb, long long sB, int transB,
         const float* __restrict__ bias, long long sBias,
         float scale,
         float* __restrict__ C, int ldc, long long sC,
         _Float16* __restrict__ C16, int ldc16, long long sC16,
         int M, int N, int K) {
  __shared__ v16h asv[32];
  __shared__ v16h bsv[32];
  _Float16* as = (_Float16*)asv;
  _Float16* bs = (_Float16*)bsv;

  int bt = blockIdx.z;
  A += (size_t)bt * sA;
  B += (size_t)bt * sB;
  const float* bb = bias ? (bias + (size_t)bt * sBias) : nullptr;

  int mt = blockIdx.x, ntile = blockIdx.y, lane = threadIdx.x;
  int n_lo = lane & 15, hi = lane >> 4;
  int n  = ntile * 16 + n_lo;
  int ma = mt * 16 + n_lo;                 // A-fragment row for this lane
  float bv0 = (bb && n < N) ? bb[n] : 0.f;
  v8f acc = {bv0, bv0, bv0, bv0, bv0, bv0, bv0, bv0};

  int KC = (K + 31) >> 5;
  for (int kc = 0; kc < KC; ++kc) {
    int kb = kc << 5;
#pragma unroll
    for (int i = 0; i < 16; ++i) {         // stage A fragment
      int k = kb + kmapA(i, hi);
      _Float16 v = (_Float16)0.f;
      if (ma < M && k < K) v = A[(size_t)ma * lda + k];
      as[lane * 16 + i] = v;
    }
#pragma unroll
    for (int i = 0; i < 16; ++i) {         // stage B fragment
      int k = kb + kmapA(i, hi);
      _Float16 v = (_Float16)0.f;
      if (n < N && k < K)
        v = transB ? B[(size_t)n * ldb + k] : B[(size_t)k * ldb + n];
      bs[lane * 16 + i] = v;
    }
    acc = wmma_f16(asv[lane], bsv[lane], acc);
  }
#pragma unroll
  for (int r = 0; r < 8; ++r) acc[r] *= scale;
#pragma unroll
  for (int r = 0; r < 8; ++r) {
    int m = mt * 16 + r + 8 * hi;
    if (m < M && n < N) {
      if (C)   C[(size_t)bt * sC + (size_t)m * ldc + n] = acc[r];
      if (C16) C16[(size_t)bt * sC16 + (size_t)m * ldc16 + n] = (_Float16)acc[r];
    }
  }
}

// ---------------------------------------------------------------------------
// Column softmax: attn[h,s,t] = softmax over s of sim[h,:,t]  (ref axis=1)
// ---------------------------------------------------------------------------
__global__ void k_softmax_col(const float* __restrict__ sim, _Float16* __restrict__ attn) {
  int t = blockIdx.x, h = blockIdx.y, tid = threadIdx.x;
  __shared__ float red[256];
  const float* base = sim + (size_t)h * N_SENT * N_SENT;
  float mx = -3.4e38f;
  for (int s = tid; s < N_SENT; s += 256) mx = fmaxf(mx, base[(size_t)s * N_SENT + t]);
  red[tid] = mx; __syncthreads();
  for (int off = 128; off > 0; off >>= 1) {
    if (tid < off) red[tid] = fmaxf(red[tid], red[tid + off]);
    __syncthreads();
  }
  mx = red[0]; __syncthreads();
  float sum = 0.f;
  for (int s = tid; s < N_SENT; s += 256) sum += __expf(base[(size_t)s * N_SENT + t] - mx);
  red[tid] = sum; __syncthreads();
  for (int off = 128; off > 0; off >>= 1) {
    if (tid < off) red[tid] += red[tid + off];
    __syncthreads();
  }
  float inv = 1.f / red[0];
  _Float16* abase = attn + (size_t)h * N_SENT * N_SENT;
  for (int s = tid; s < N_SENT; s += 256)
    abase[(size_t)s * N_SENT + t] = (_Float16)(__expf(base[(size_t)s * N_SENT + t] - mx) * inv);
}

// high_sent_feature[r] = feat_w[r,:] . multi_flat + feat_b[r]   (bandwidth bound)
__global__ void k_featvec(const float* __restrict__ fw, const float* __restrict__ fb,
                          const float* __restrict__ multi, float* __restrict__ out) {
  int r = blockIdx.x, tid = threadIdx.x;
  __shared__ float red[256];
  const float* row = fw + (size_t)r * (N_SENT * DMODEL);
  float s = 0.f;
  for (int i = tid; i < N_SENT * DMODEL; i += 256) s += row[i] * multi[i];
  red[tid] = s; __syncthreads();
  for (int off = 128; off > 0; off >>= 1) {
    if (tid < off) red[tid] += red[tid + off];
    __syncthreads();
  }
  if (tid == 0) out[r] = red[0] + fb[r];
}

__global__ void k_copy_head(const float* __restrict__ sent_enc, float* __restrict__ out) {
  int i = blockIdx.x * 256 + threadIdx.x;
  if (i < DMODEL) out[DMODEL + i] = sent_enc[(size_t)N_SENT * DMODEL + i];
}

// TextCNN conv (padding w-1 both sides) + max-pool + bias; one block per filter
__global__ void k_conv_pool(const float* __restrict__ x, const float* __restrict__ cw,
                            const float* __restrict__ conv_b, float* __restrict__ out,
                            int w, int j) {
  int f = blockIdx.x, tid = threadIdx.x;
  __shared__ float red[256];
  int P = N_SENT + w - 1;
  float mx = -3.4e38f;
  for (int p = tid; p < P; p += 256) {
    float s = 0.f;
    for (int r = 0; r < w; ++r) {
      int si = p - (w - 1) + r;
      if (si >= 0 && si < N_SENT) {
        const float* xr = x + (size_t)si * DMODEL;
        const float* wr = cw + ((size_t)f * w + r) * DMODEL;
        for (int d = 0; d < DMODEL; ++d) s += xr[d] * wr[d];
      }
    }
    mx = fmaxf(mx, s);
  }
  red[tid] = mx; __syncthreads();
  for (int off = 128; off > 0; off >>= 1) {
    if (tid < off) red[tid] = fmaxf(red[tid], red[tid + off]);
    __syncthreads();
  }
  if (tid == 0) out[f] = red[0] + conv_b[j * 10 + f];
}

__global__ void k_f32_to_f16(const float* __restrict__ in, _Float16* __restrict__ outp,
                             long long n) {
  long long i = (long long)blockIdx.x * blockDim.x + threadIdx.x;
  if (i < n) outp[i] = (_Float16)in[i];
}

// ---------------------------------------------------------------------------
extern "C" void kernel_launch(void* const* d_in, const int* in_sizes, int n_in,
                              void* d_out, int out_size, void* d_ws, size_t ws_size,
                              hipStream_t stream) {
  (void)in_sizes; (void)n_in; (void)out_size; (void)ws_size;
  const int*   head_tok = (const int*)d_in[0];
  const int*   body_tok = (const int*)d_in[1];
  const float* emb      = (const float*)d_in[2];
  const float* wih_f = (const float*)d_in[3];
  const float* whh_f = (const float*)d_in[4];
  const float* bih_f = (const float*)d_in[5];
  const float* bhh_f = (const float*)d_in[6];
  const float* wih_b = (const float*)d_in[7];
  const float* whh_b = (const float*)d_in[8];
  const float* bih_b = (const float*)d_in[9];
  const float* bhh_b = (const float*)d_in[10];
  const float* wq = (const float*)d_in[11];
  const float* bq = (const float*)d_in[12];
  const float* wk = (const float*)d_in[13];
  const float* bk = (const float*)d_in[14];
  const float* wv = (const float*)d_in[15];
  const float* bvv = (const float*)d_in[16];
  const float* concat_w = (const float*)d_in[17];
  const float* concat_b = (const float*)d_in[18];
  const float* feat_w = (const float*)d_in[19];
  const float* feat_b = (const float*)d_in[20];
  const float* conv_b = (const float*)d_in[21];
  const float* conv_w[7];
  for (int j = 0; j < 7; ++j) conv_w[j] = (const float*)d_in[22 + j];
  float* out = (float*)d_out;

  char* ws = (char*)d_ws;
  size_t off = 0;
  auto take = [&](size_t bytes) -> char* {
    char* p = ws + off;
    off = (off + bytes + 255) & ~(size_t)255;
    return p;
  };
  _Float16* xfrag  = (_Float16*)take((size_t)SEQ_TILES * N_WORD * KC_IH * 512 * 2);
  _Float16* pk_ih  = (_Float16*)take((size_t)2 * NTILES_G * KC_IH * 512 * 2);
  _Float16* pk_hh  = (_Float16*)take((size_t)2 * NTILES_G * KC_HH * 512 * 2);
  float*    bias_c = (float*)take((size_t)2 * NGATE * 4);
  float*    sent_enc = (float*)take((size_t)(SEQ_TILES * 16) * DMODEL * 4);
  _Float16* sent16 = (_Float16*)take((size_t)N_SENT * DMODEL * 2);
  _Float16* wq16 = (_Float16*)take((size_t)N_HEAD * DMODEL * DMODEL * 2);
  _Float16* wk16 = (_Float16*)take((size_t)N_HEAD * DMODEL * DMODEL * 2);
  _Float16* wv16 = (_Float16*)take((size_t)N_HEAD * DMODEL * DMODEL * 2);
  _Float16* cw16 = (_Float16*)take((size_t)DMODEL * N_HEAD * DMODEL * 2);
  _Float16* Q16 = (_Float16*)take((size_t)N_HEAD * N_SENT * DMODEL * 2);
  _Float16* K16 = (_Float16*)take((size_t)N_HEAD * N_SENT * DMODEL * 2);
  _Float16* V16 = (_Float16*)take((size_t)N_HEAD * N_SENT * DMODEL * 2);
  float*    sim = (float*)take((size_t)N_HEAD * N_SENT * N_SENT * 4);
  _Float16* attn16 = (_Float16*)take((size_t)N_HEAD * N_SENT * N_SENT * 2);
  float*    HC = (float*)take((size_t)N_SENT * N_HEAD * DMODEL * 4);
  _Float16* HC16 = (_Float16*)take((size_t)N_SENT * N_HEAD * DMODEL * 2);
  float*    multi = (float*)take((size_t)N_SENT * DMODEL * 4);

  // ---- weight packing + embedding gather into WMMA fragment layouts ----
  {
    size_t tot = (size_t)2 * NTILES_G * KC_IH * 512;
    k_pack_w<<<dim3((unsigned)((tot + 255) / 256)), dim3(256), 0, stream>>>(
        wih_f, wih_b, pk_ih, IN_DIM, KC_IH);
  }
  {
    size_t tot = (size_t)2 * NTILES_G * KC_HH * 512;
    k_pack_w<<<dim3((unsigned)((tot + 255) / 256)), dim3(256), 0, stream>>>(
        whh_f, whh_b, pk_hh, MEM, KC_HH);
  }
  k_bias<<<dim3(4), dim3(256), 0, stream>>>(bih_f, bhh_f, bih_b, bhh_b, bias_c);
  {
    size_t tot = (size_t)SEQ_TILES * N_WORD * KC_IH * 512;
    k_gather<<<dim3((unsigned)((tot + 255) / 256)), dim3(256), 0, stream>>>(
        head_tok, body_tok, emb, xfrag);
  }

  // ---- batched BiLSTM over all 513 sequences (dominant WMMA work) ----
  k_lstm<<<dim3(SEQ_TILES, 2), dim3(32), 0, stream>>>(xfrag, pk_ih, pk_hh, bias_c, sent_enc);

  // ---- f16 conversions for attention GEMMs ----
  k_f32_to_f16<<<dim3((N_SENT * DMODEL + 255) / 256), 256, 0, stream>>>(
      sent_enc, sent16, (long long)N_SENT * DMODEL);
  k_f32_to_f16<<<dim3((N_HEAD * DMODEL * DMODEL + 255) / 256), 256, 0, stream>>>(
      wq, wq16, (long long)N_HEAD * DMODEL * DMODEL);
  k_f32_to_f16<<<dim3((N_HEAD * DMODEL * DMODEL + 255) / 256), 256, 0, stream>>>(
      wk, wk16, (long long)N_HEAD * DMODEL * DMODEL);
  k_f32_to_f16<<<dim3((N_HEAD * DMODEL * DMODEL + 255) / 256), 256, 0, stream>>>(
      wv, wv16, (long long)N_HEAD * DMODEL * DMODEL);
  k_f32_to_f16<<<dim3((DMODEL * N_HEAD * DMODEL + 255) / 256), 256, 0, stream>>>(
      concat_w, cw16, (long long)DMODEL * N_HEAD * DMODEL);

  // ---- Q/K/V projections, batched over 8 heads (A shared via stride 0) ----
  dim3 gqkv(32, 13, N_HEAD);
  k_gemm16<<<gqkv, 32, 0, stream>>>(sent16, DMODEL, 0LL,
                                    wq16, DMODEL, (long long)DMODEL * DMODEL, 1,
                                    bq, (long long)DMODEL, 1.f,
                                    (float*)nullptr, 0, 0LL,
                                    Q16, DMODEL, (long long)N_SENT * DMODEL,
                                    N_SENT, DMODEL, DMODEL);
  k_gemm16<<<gqkv, 32, 0, stream>>>(sent16, DMODEL, 0LL,
                                    wk16, DMODEL, (long long)DMODEL * DMODEL, 1,
                                    bk, (long long)DMODEL, 1.f,
                                    (float*)nullptr, 0, 0LL,
                                    K16, DMODEL, (long long)N_SENT * DMODEL,
                                    N_SENT, DMODEL, DMODEL);
  k_gemm16<<<gqkv, 32, 0, stream>>>(sent16, DMODEL, 0LL,
                                    wv16, DMODEL, (long long)DMODEL * DMODEL, 1,
                                    bvv, (long long)DMODEL, 1.f,
                                    (float*)nullptr, 0, 0LL,
                                    V16, DMODEL, (long long)N_SENT * DMODEL,
                                    N_SENT, DMODEL, DMODEL);

  // ---- sim = Q @ K^T / sqrt(200) ----
  dim3 gsim(32, 32, N_HEAD);
  k_gemm16<<<gsim, 32, 0, stream>>>(Q16, DMODEL, (long long)N_SENT * DMODEL,
                                    K16, DMODEL, (long long)N_SENT * DMODEL, 1,
                                    (const float*)nullptr, 0LL, 0.07071067812f,
                                    sim, N_SENT, (long long)N_SENT * N_SENT,
                                    (_Float16*)nullptr, 0, 0LL,
                                    N_SENT, N_SENT, DMODEL);

  // ---- softmax over s (ref axis=1) ----
  k_softmax_col<<<dim3(N_SENT, N_HEAD), 256, 0, stream>>>(sim, attn16);

  // ---- out = attn @ V, written head-concatenated into HC [512,1600] ----
  dim3 gout(32, 13, N_HEAD);
  k_gemm16<<<gout, 32, 0, stream>>>(attn16, N_SENT, (long long)N_SENT * N_SENT,
                                    V16, DMODEL, (long long)N_SENT * DMODEL, 0,
                                    (const float*)nullptr, 0LL, 1.f,
                                    HC, N_HEAD * DMODEL, (long long)DMODEL,
                                    HC16, N_HEAD * DMODEL, (long long)DMODEL,
                                    N_SENT, DMODEL, N_SENT);

  // ---- multi = HC @ concat_w^T + concat_b ----
  dim3 gmul(32, 13, 1);
  k_gemm16<<<gmul, 32, 0, stream>>>(HC16, N_HEAD * DMODEL, 0LL,
                                    cw16, N_HEAD * DMODEL, 0LL, 1,
                                    concat_b, 0LL, 1.f,
                                    multi, DMODEL, 0LL,
                                    (_Float16*)nullptr, 0, 0LL,
                                    N_SENT, DMODEL, N_HEAD * DMODEL);

  // ---- outputs ----
  k_featvec<<<dim3(DMODEL), 256, 0, stream>>>(feat_w, feat_b, multi, out);   // [0..200)
  k_copy_head<<<dim3(1), 256, 0, stream>>>(sent_enc, out);                   // [200..400)
  for (int j = 0; j < 7; ++j)                                                // [400..470)
    k_conv_pool<<<dim3(10), 256, 0, stream>>>(sent_enc, conv_w[j], conv_b,
                                              out + 2 * DMODEL + j * 10, j + 1, j);
}
